// AttentionHead_45019847196892
// MI455X (gfx1250) — compile-verified, matrix operand order checked by reading, not run on hardware
//
#include <hip/hip_runtime.h>

typedef __attribute__((ext_vector_type(16))) __bf16 v16bf;
typedef __attribute__((ext_vector_type(8)))  float  v8f;

#define S_LEN   2048
#define D_MODEL 1024
#define D_HEAD  128
#define NBATCH  4
#define KV_TILE 64
#define PL_STRIDE 72    // 64 + 8 pad (ushorts)

struct alignas(16) U4 { unsigned int x, y, z, w; };
struct alignas(16) F4 { float x, y, z, w; };

union Frag {
  v16bf v;
  __bf16 h[16];
  unsigned short u[16];
};

static __device__ __forceinline__ v8f wmma_bf16(const Frag& a, const Frag& b, v8f c) {
  // (neg_a, A, neg_b, B, c_mod, C, reuse_a, reuse_b)
  return __builtin_amdgcn_wmma_f32_16x16x32_bf16(false, a.v, false, b.v, (short)0, c, false, false);
}

static __device__ __forceinline__ void load_frag(Frag& f, const unsigned short* p) {
  *(U4*)&f.u[0] = *(const U4*)p;
  *(U4*)&f.u[8] = *(const U4*)(p + 16);
}

// ---------------------------------------------------------------------------
// Kernel 0: transpose + bf16-convert weights. Wq gets 1/sqrt(128) folded in.
// Wxt layout: [n (128)][k (1024)]   (rows = output column -> B-fragment rows)
// Wot layout: [n (1024)][k (128)]
// ---------------------------------------------------------------------------
__global__ void prep_weights(const float* __restrict__ Wq, const float* __restrict__ Wk,
                             const float* __restrict__ Wv, const float* __restrict__ Wo,
                             unsigned short* __restrict__ Wqt, unsigned short* __restrict__ Wkt,
                             unsigned short* __restrict__ Wvt, unsigned short* __restrict__ Wot) {
  const float qs = 0.08838834764831845f;  // 1/sqrt(128)
  __bf16* wq = (__bf16*)Wqt;
  __bf16* wk = (__bf16*)Wkt;
  __bf16* wv = (__bf16*)Wvt;
  __bf16* wo = (__bf16*)Wot;
  for (int idx = blockIdx.x * blockDim.x + threadIdx.x; idx < D_MODEL * D_HEAD;
       idx += gridDim.x * blockDim.x) {
    int n = idx & (D_HEAD - 1);
    int k = idx >> 7;
    wq[n * D_MODEL + k] = (__bf16)(Wq[k * D_HEAD + n] * qs);
    wk[n * D_MODEL + k] = (__bf16)Wk[k * D_HEAD + n];
    wv[n * D_MODEL + k] = (__bf16)Wv[k * D_HEAD + n];
    // Wot[no*128 + ko] = Wo[ko*1024 + no], with no = idx>>7, ko = idx&127
    wo[idx] = (__bf16)Wo[(idx & (D_HEAD - 1)) * D_MODEL + (idx >> 7)];
  }
}

// ---------------------------------------------------------------------------
// Kernel 1: Q/K/V projection. GEMM [8192 x 128] = enc[8192 x 1024] * W[1024 x 128]
// grid (128, 3): x -> 4 waves * 16-row strips, y -> which projection.
// Q, K stored row-major [B*S][128]; V stored TRANSPOSED [B][128][S] so the
// attention kernel can read V B-fragments directly from global with b128 loads.
// All 8 B fragments are materialized before the 8 WMMAs so loads pipeline.
// ---------------------------------------------------------------------------
__global__ void qkv_proj(const float* __restrict__ enc,
                         const unsigned short* __restrict__ Wqt,
                         const unsigned short* __restrict__ Wkt,
                         const unsigned short* __restrict__ Wvt,
                         unsigned short* __restrict__ Qb,
                         unsigned short* __restrict__ Kb,
                         unsigned short* __restrict__ Vb) {
  const unsigned short* Wt = (blockIdx.y == 0) ? Wqt : (blockIdx.y == 1) ? Wkt : Wvt;
  unsigned short* dst      = (blockIdx.y == 0) ? Qb  : (blockIdx.y == 1) ? Kb  : Vb;
  int wave = threadIdx.x >> 5, lane = threadIdx.x & 31;
  int row = lane & 15, hi = lane >> 4, kg = hi * 8;
  int m0 = (blockIdx.x * 4 + wave) * 16;

  const float* arow = enc + (size_t)(m0 + row) * D_MODEL;
  v8f acc[8];
  for (int n = 0; n < 8; ++n) acc[n] = {0.f,0.f,0.f,0.f,0.f,0.f,0.f,0.f};

#pragma unroll 2
  for (int kb = 0; kb < D_MODEL; kb += 32) {
    Frag a;
    F4 f0 = *(const F4*)(arow + kb + kg);
    F4 f1 = *(const F4*)(arow + kb + kg + 4);
    F4 f2 = *(const F4*)(arow + kb + 16 + kg);
    F4 f3 = *(const F4*)(arow + kb + 16 + kg + 4);
    a.h[0]=(__bf16)f0.x;  a.h[1]=(__bf16)f0.y;  a.h[2]=(__bf16)f0.z;  a.h[3]=(__bf16)f0.w;
    a.h[4]=(__bf16)f1.x;  a.h[5]=(__bf16)f1.y;  a.h[6]=(__bf16)f1.z;  a.h[7]=(__bf16)f1.w;
    a.h[8]=(__bf16)f2.x;  a.h[9]=(__bf16)f2.y;  a.h[10]=(__bf16)f2.z; a.h[11]=(__bf16)f2.w;
    a.h[12]=(__bf16)f3.x; a.h[13]=(__bf16)f3.y; a.h[14]=(__bf16)f3.z; a.h[15]=(__bf16)f3.w;
    if (kb + 32 < D_MODEL) __builtin_prefetch(arow + kb + 32, 0, 0);  // global_prefetch

    Frag b[8];
    for (int n = 0; n < 8; ++n)
      load_frag(b[n], Wt + (size_t)(n * 16 + row) * D_MODEL + kb + kg);
    for (int n = 0; n < 8; ++n)
      acc[n] = wmma_bf16(a, b[n], acc[n]);
  }

  if (blockIdx.y == 2) {
    // V: transposed store [b][col (128)][s (2048)]; C-layout rows are contiguous
    int bidx = m0 >> 11;           // batch
    int s0   = m0 & (S_LEN - 1);   // seq offset within batch
    for (int n = 0; n < 8; ++n) {
      alignas(16) __bf16 tmp[8];
      for (int j = 0; j < 8; ++j) tmp[j] = (__bf16)acc[n][j];
      size_t off = ((size_t)bidx * D_HEAD + n * 16 + row) * S_LEN + s0 + 8 * hi;
      *(U4*)(dst + off) = *(const U4*)tmp;
    }
  } else {
    __bf16* d = (__bf16*)dst;
    for (int n = 0; n < 8; ++n)
      for (int j = 0; j < 8; ++j)
        d[(size_t)(m0 + j + 8 * hi) * D_HEAD + n * 16 + row] = (__bf16)acc[n][j];
  }
}

// ---------------------------------------------------------------------------
// Kernel 2: fused flash attention (causal, online softmax).
// grid (16, 4): x -> 128-row q block (8 waves x 16 rows), y -> batch. block=256.
// K and V(transposed) B-fragments are read straight from global (L2-resident:
// 512 KB per batch each). Waves are fully independent -- no block barriers.
// Only P staging round-trips through per-wave LDS (C-layout -> A-fragment).
// ---------------------------------------------------------------------------
__global__ void attention(const unsigned short* __restrict__ Qb,
                          const unsigned short* __restrict__ Kb,
                          const unsigned short* __restrict__ Vt,
                          unsigned short* __restrict__ Sc) {
  __shared__ unsigned short Pl[8][16 * PL_STRIDE];

  int wave = threadIdx.x >> 5, lane = threadIdx.x & 31;
  int row = lane & 15, hi = lane >> 4, kg = hi * 8;
  int bb = blockIdx.y;
  int q0 = blockIdx.x * 128 + wave * 16;
  size_t bbase  = (size_t)bb * S_LEN * D_HEAD;      // Q/K/Sc row-major base
  size_t vtbase = (size_t)bb * D_HEAD * S_LEN;      // V transposed base

  // Q fragments resident in registers (1/sqrt(d) already folded into Wq)
  Frag qf[4];
  {
    const unsigned short* qrow = Qb + bbase + (size_t)(q0 + row) * D_HEAD;
    for (int kc = 0; kc < 4; ++kc)
      load_frag(qf[kc], qrow + kc * 32 + kg);
  }

  v8f o[8];
  for (int n = 0; n < 8; ++n) o[n] = {0.f,0.f,0.f,0.f,0.f,0.f,0.f,0.f};
  float mrow[8], lrow[8];
  for (int j = 0; j < 8; ++j) { mrow[j] = -3.0e38f; lrow[j] = 0.f; }

  int nkv = (q0 >> 6) + 1;  // causal: key blocks 0 .. q0/64
  for (int kv = 0; kv < nkv; ++kv) {
    // S = Q * K^T  (4 key sub-tiles x 4 K-dim chunks), K read from global.
    // Batch the 4 K-fragments per sub-tile so loads overlap the WMMAs.
    const unsigned short* Kblk = Kb + bbase + (size_t)kv * KV_TILE * D_HEAD;
    if (kv + 1 < nkv)
      __builtin_prefetch(Kb + bbase + (size_t)(kv + 1) * KV_TILE * D_HEAD + row * D_HEAD, 0, 0);
    v8f s[4];
    for (int t = 0; t < 4; ++t) {
      Frag b[4];
      for (int kc = 0; kc < 4; ++kc)
        load_frag(b[kc], Kblk + (size_t)(t * 16 + row) * D_HEAD + kc * 32 + kg);
      s[t] = {0.f,0.f,0.f,0.f,0.f,0.f,0.f,0.f};
      for (int kc = 0; kc < 4; ++kc)
        s[t] = wmma_bf16(qf[kc], b[kc], s[t]);
    }

    // causal mask + online softmax (branch-free; 16-lane group shuffles)
    float pv[4][8];
    for (int j = 0; j < 8; ++j) {
      int qi = q0 + j + 8 * hi;
      float mx = -3.0e38f;
      for (int t = 0; t < 4; ++t) {
        int ki = kv * KV_TILE + t * 16 + row;
        float v = (ki > qi) ? -1.0e9f : s[t][j];
        pv[t][j] = v;
        mx = fmaxf(mx, v);
      }
      mx = fmaxf(mx, __shfl_xor(mx, 1));
      mx = fmaxf(mx, __shfl_xor(mx, 2));
      mx = fmaxf(mx, __shfl_xor(mx, 4));
      mx = fmaxf(mx, __shfl_xor(mx, 8));
      float mnew  = fmaxf(mrow[j], mx);
      float alpha = __expf(mrow[j] - mnew);
      mrow[j] = mnew;
      float rs = 0.f;
      for (int t = 0; t < 4; ++t) {
        float p = __expf(pv[t][j] - mnew);
        pv[t][j] = p;
        rs += p;
      }
      rs += __shfl_xor(rs, 1);
      rs += __shfl_xor(rs, 2);
      rs += __shfl_xor(rs, 4);
      rs += __shfl_xor(rs, 8);
      lrow[j] = lrow[j] * alpha + rs;
      for (int n = 0; n < 8; ++n) o[n][j] *= alpha;
      // stage P (C-layout -> memory-layout) in per-wave LDS
      __bf16* prow = (__bf16*)&Pl[wave][(j + 8 * hi) * PL_STRIDE];
      for (int t = 0; t < 4; ++t)
        prow[t * 16 + row] = (__bf16)pv[t][j];
    }
    asm volatile("s_wait_dscnt 0" ::: "memory");  // same-wave LDS RAW

    // O += P * V   (V^T read from global: lane n gets contiguous key run).
    // Materialize all 8 V-fragments per key-chunk before the 8 WMMAs.
    for (int kc = 0; kc < 2; ++kc) {
      Frag pf;
      load_frag(pf, &Pl[wave][row * PL_STRIDE + kc * 32 + kg]);
      Frag b[8];
      for (int n = 0; n < 8; ++n)
        load_frag(b[n],
                  Vt + vtbase + (size_t)(n * 16 + row) * S_LEN + kv * KV_TILE + kc * 32 + kg);
      for (int n = 0; n < 8; ++n)
        o[n] = wmma_bf16(pf, b[n], o[n]);
    }
  }

  float inv[8];
  for (int j = 0; j < 8; ++j) inv[j] = 1.f / lrow[j];
  __bf16* sc = (__bf16*)Sc;
  for (int n = 0; n < 8; ++n)
    for (int j = 0; j < 8; ++j)
      sc[bbase + (size_t)(q0 + j + 8 * hi) * D_HEAD + n * 16 + row] =
          (__bf16)(o[n][j] * inv[j]);
}

// ---------------------------------------------------------------------------
// Kernel 3: output projection. [8192 x 1024] = Sc[8192 x 128] * Wo[128 x 1024]
// grid (128, 8): x -> 4 waves x 16-row strips, y -> 128-col chunk.
// ---------------------------------------------------------------------------
__global__ void out_proj(const unsigned short* __restrict__ Sc,
                         const unsigned short* __restrict__ Wot,
                         float* __restrict__ out) {
  int wave = threadIdx.x >> 5, lane = threadIdx.x & 31;
  int row = lane & 15, hi = lane >> 4, kg = hi * 8;
  int m0 = (blockIdx.x * 4 + wave) * 16;
  int n0 = blockIdx.y * 128;

  v8f acc[8];
  for (int n = 0; n < 8; ++n) acc[n] = {0.f,0.f,0.f,0.f,0.f,0.f,0.f,0.f};
  const unsigned short* arow = Sc + (size_t)(m0 + row) * D_HEAD;

  for (int kc = 0; kc < 4; ++kc) {
    Frag a;
    load_frag(a, arow + kc * 32 + kg);
    Frag b[8];
    for (int n = 0; n < 8; ++n)
      load_frag(b[n], Wot + (size_t)(n0 + n * 16 + row) * D_HEAD + kc * 32 + kg);
    for (int n = 0; n < 8; ++n)
      acc[n] = wmma_bf16(a, b[n], acc[n]);
  }
  for (int n = 0; n < 8; ++n)
    for (int j = 0; j < 8; ++j)
      out[(size_t)(m0 + j + 8 * hi) * D_MODEL + n0 + n * 16 + row] = acc[n][j];
}

// ---------------------------------------------------------------------------
extern "C" void kernel_launch(void* const* d_in, const int* in_sizes, int n_in,
                              void* d_out, int out_size, void* d_ws, size_t ws_size,
                              hipStream_t stream) {
  (void)in_sizes; (void)n_in; (void)out_size; (void)ws_size;
  const float* enc = (const float*)d_in[0];
  // d_in[1] is the causal mask: structure is known, recomputed in-kernel.
  const float* Wq = (const float*)d_in[2];
  const float* Wk = (const float*)d_in[3];
  const float* Wv = (const float*)d_in[4];
  const float* Wo = (const float*)d_in[5];
  float* out = (float*)d_out;

  const size_t QSZ = (size_t)NBATCH * S_LEN * D_HEAD * sizeof(unsigned short); // 2 MiB
  const size_t WTE = (size_t)D_MODEL * D_HEAD;                                 // 131072 elems
  char* ws = (char*)d_ws;
  unsigned short* Qb  = (unsigned short*)(ws);            // [B*S][128]
  unsigned short* Kb  = (unsigned short*)(ws + QSZ);      // [B*S][128]
  unsigned short* Vb  = (unsigned short*)(ws + 2 * QSZ);  // [B][128][S] (transposed)
  unsigned short* Sc  = (unsigned short*)(ws + 3 * QSZ);  // [B*S][128]
  unsigned short* Wqt = (unsigned short*)(ws + 4 * QSZ);
  unsigned short* Wkt = Wqt + WTE;
  unsigned short* Wvt = Wkt + WTE;
  unsigned short* Wot = Wvt + WTE;

  prep_weights<<<512, 256, 0, stream>>>(Wq, Wk, Wv, Wo, Wqt, Wkt, Wvt, Wot);
  qkv_proj<<<dim3(128, 3), 128, 0, stream>>>(enc, Wqt, Wkt, Wvt, Qb, Kb, Vb);
  attention<<<dim3(16, NBATCH), 256, 0, stream>>>(Qb, Kb, Vb, Sc);
  out_proj<<<dim3(128, 8), 128, 0, stream>>>(Sc, Wot, out);
}
